// SimNet_35416300323447
// MI455X (gfx1250) — matrix operand order repaired and checked by reading, not compiled, not run on hardware
//
#include <hip/hip_runtime.h>

// ---------------------------------------------------------------------------
// SimNet: sims = <stance, body_n>  -> top-5 gather -> concat -> 5-layer MLP.
// Stage 1 (805 MB stream) is bandwidth-bound: coalesced float4 NT loads.
// Stage 3 (MLP) uses v_wmma_f32_16x16x32_f16 (f16 inputs, f32 accumulate).
// ---------------------------------------------------------------------------

#define BATCH 128
#define NSENT 4096
#define DIM   384
#define TOPK  5
#define XDIM  ((TOPK + 1) * DIM)   // 2304
#define H1DIM 1024
#define H2DIM 512
#define CDIM  2

typedef __attribute__((ext_vector_type(4)))  float     v4f;
typedef __attribute__((ext_vector_type(8)))  float     v8f;
typedef __attribute__((ext_vector_type(8)))  _Float16  v8h;
typedef __attribute__((ext_vector_type(16))) _Float16  v16h;

// ---------------------------------------------------------------------------
// Kernel 1: sims[b, n] = dot(stance[b], body[b, n])
// Grid: 128 * 16 blocks, 256 threads (8 waves). Each block covers 256 rows of
// one batch element; each wave computes one full 384-dot per row iteration.
// body is streamed once (805 MB) -> non-temporal loads keep it out of L2.
// ---------------------------------------------------------------------------
__global__ void sims_kernel(const float* __restrict__ stance,
                            const float* __restrict__ body,
                            float* __restrict__ sims) {
  const int b     = blockIdx.x >> 4;
  const int chunk = blockIdx.x & 15;

  __shared__ __align__(16) float st[DIM];
  for (int i = threadIdx.x; i < DIM; i += blockDim.x)
    st[i] = stance[b * DIM + i];
  __syncthreads();

  const int wave = threadIdx.x >> 5;
  const int lane = threadIdx.x & 31;

  for (int r = wave; r < 256; r += 8) {
    const int n = chunk * 256 + r;
    const float* row = body + ((size_t)b * NSENT + n) * DIM;
    float acc = 0.0f;
#pragma unroll
    for (int j = 0; j < 3; ++j) {            // 3 * 32 lanes * 4 floats = 384
      const int o = j * 128 + lane * 4;
      v4f v = __builtin_nontemporal_load((const v4f*)(row + o));
      v4f s = *(const v4f*)(&st[o]);
      acc += v.x * s.x + v.y * s.y + v.z * s.z + v.w * s.w;
    }
#pragma unroll
    for (int off = 16; off; off >>= 1) acc += __shfl_xor(acc, off, 32);
    if (lane == 0) sims[(size_t)b * NSENT + n] = acc;
  }
}

// ---------------------------------------------------------------------------
// Kernel 2: per-batch top-5 (jax tie-break: lowest index wins) + concat.
// Writes x[b] = [stance(384) | top0..top4 (5*384)] as f16 for the WMMA MLP.
// ---------------------------------------------------------------------------
__global__ void topk_concat_kernel(const float* __restrict__ stance,
                                   const float* __restrict__ body,
                                   const float* __restrict__ sims,
                                   _Float16* __restrict__ xh) {
  const int b   = blockIdx.x;
  const int tid = threadIdx.x;

  __shared__ float sl[NSENT];
  __shared__ float rv[256];
  __shared__ int   ri[256];
  __shared__ int   topIdx[TOPK];

  for (int i = tid; i < NSENT; i += 256) sl[i] = sims[(size_t)b * NSENT + i];
  __syncthreads();

  for (int k = 0; k < TOPK; ++k) {
    float best = -3.4e38f;
    int   bi   = 0x7fffffff;
    for (int i = tid; i < NSENT; i += 256) {
      float v = sl[i];
      if (v > best || (v == best && i < bi)) { best = v; bi = i; }
    }
    rv[tid] = best; ri[tid] = bi;
    __syncthreads();
    for (int s = 128; s > 0; s >>= 1) {
      if (tid < s) {
        float vo = rv[tid + s]; int io = ri[tid + s];
        if (vo > rv[tid] || (vo == rv[tid] && io < ri[tid])) {
          rv[tid] = vo; ri[tid] = io;
        }
      }
      __syncthreads();
    }
    if (tid == 0) { topIdx[k] = ri[0]; sl[ri[0]] = -3.4e38f; }
    __syncthreads();
  }

  _Float16* xrow = xh + (size_t)b * XDIM;
  for (int i = tid; i < DIM; i += 256)
    xrow[i] = (_Float16)stance[b * DIM + i];
  for (int k = 0; k < TOPK; ++k) {
    const float* src = body + ((size_t)b * NSENT + topIdx[k]) * DIM;
    for (int i = tid; i < DIM; i += 256)
      xrow[DIM + k * DIM + i] = (_Float16)src[i];
  }
}

// ---------------------------------------------------------------------------
// fp32 -> fp16 weight conversion (weights live in L2 afterwards: ~10 MB)
// ---------------------------------------------------------------------------
__global__ void f32_to_f16_kernel(const float* __restrict__ src,
                                  _Float16* __restrict__ dst, int n) {
  int i = blockIdx.x * blockDim.x + threadIdx.x;
  if (i < n) dst[i] = (_Float16)src[i];
}

// ---------------------------------------------------------------------------
// WMMA linear layer: Out = act(A[M,Kd] @ W^T + bias),  W row-major [Nout,Kd].
// One wave per 16x16 output tile; K-loop in steps of 32 with
// v_wmma_f32_16x16x32_f16 (fp32 accumulators).
// Fragment addressing (this code's k order, the same for A and B):
//   A (16x32): lane l -> m=l&15; K run [(l>>4)*16,+16) contiguous
//              => two contiguous 16B loads, shuffled into one v16h.
//   B (32x16): lane l -> n=l&15; K run [(l>>4)*16,+16) contiguous
//              => one 32B load straight from row n of W (B = W^T).
//   C/D: VGPR r -> m = m0 + (l>>4)*8 + r, n = n0 + (l&15).
// ---------------------------------------------------------------------------
__global__ void wmma_linear_kernel(const _Float16* __restrict__ A,
                                   const _Float16* __restrict__ W,
                                   const float* __restrict__ bias,
                                   _Float16* __restrict__ Out,
                                   int M, int Kd, int Nout, int relu) {
  const int wave = (blockIdx.x * blockDim.x + threadIdx.x) >> 5;
  const int lane = threadIdx.x & 31;
  const int tilesM = M >> 4;
  const int tm = wave % tilesM;
  const int tn = wave / tilesM;
  if (tn * 16 >= Nout) return;           // wave-uniform: EXEC stays all-ones

  const int m0 = tm * 16, n0 = tn * 16;
  const int la = lane & 15, lh = lane >> 4;

  const _Float16* pA = A + (size_t)(m0 + la) * Kd + lh * 16;
  const _Float16* pB = W + (size_t)(n0 + la) * Kd + lh * 16;

  v8f acc = {};
  for (int kk = 0; kk < Kd; kk += 32) {
    v8h  alo = *(const v8h*)(pA + kk);        // K = kb .. kb+7
    v8h  ahi = *(const v8h*)(pA + kk + 8);    // K = kb+8 .. kb+15
    v16h a   = __builtin_shufflevector(alo, ahi,
                 0, 1, 2, 3, 4, 5, 6, 7, 8, 9, 10, 11, 12, 13, 14, 15);
    v16h bf  = *(const v16h*)(pB + kk);       // K = kb2 .. kb2+15
    acc = __builtin_amdgcn_wmma_f32_16x16x32_f16(
        /*neg_a=*/false, a, /*neg_b=*/false, bf,
        /*c_mod=*/(short)0, acc, /*reuse_a=*/false, /*reuse_b=*/false);
  }

  const int n = n0 + la;
  const float bv = bias[n];
#pragma unroll
  for (int r = 0; r < 8; ++r) {
    float v = acc[r] + bv;
    if (relu) v = v > 0.0f ? v : 0.0f;
    Out[(size_t)(m0 + lh * 8 + r) * Nout + n] = (_Float16)v;
  }
}

// ---------------------------------------------------------------------------
// Final 512 -> 2 layer in fp32 (too narrow for a WMMA tile; 256 tiny dots).
// One wave per output element.
// ---------------------------------------------------------------------------
__global__ void final_layer_kernel(const _Float16* __restrict__ H,
                                   const float* __restrict__ W5,
                                   const float* __restrict__ b5,
                                   float* __restrict__ out) {
  const int wave = (blockIdx.x * blockDim.x + threadIdx.x) >> 5;
  const int lane = threadIdx.x & 31;
  const int b = wave >> 1, c = wave & 1;
  float s = 0.0f;
  for (int d = lane; d < H2DIM; d += 32)
    s += (float)H[b * H2DIM + d] * W5[c * H2DIM + d];
#pragma unroll
  for (int off = 16; off; off >>= 1) s += __shfl_xor(s, off, 32);
  if (lane == 0) out[b * CDIM + c] = s + b5[c];
}

// ---------------------------------------------------------------------------
static inline char* carve(char*& p, size_t bytes) {
  char* r = p;
  p += (bytes + 255) & ~(size_t)255;
  return r;
}

extern "C" void kernel_launch(void* const* d_in, const int* in_sizes, int n_in,
                              void* d_out, int out_size, void* d_ws,
                              size_t ws_size, hipStream_t stream) {
  const float* stance = (const float*)d_in[0];   // sim_stance_emb [128,384]
  const float* body   = (const float*)d_in[2];   // sim_body_emb [128,4096,384]
  const float* W1 = (const float*)d_in[4];  const float* b1 = (const float*)d_in[5];
  const float* W2 = (const float*)d_in[6];  const float* b2 = (const float*)d_in[7];
  const float* W3 = (const float*)d_in[8];  const float* b3 = (const float*)d_in[9];
  const float* W4 = (const float*)d_in[10]; const float* b4 = (const float*)d_in[11];
  const float* W5 = (const float*)d_in[12]; const float* b5 = (const float*)d_in[13];
  float* out = (float*)d_out;

  char* p = (char*)d_ws;
  float*    sims = (float*)   carve(p, (size_t)BATCH * NSENT * 4);
  _Float16* xh   = (_Float16*)carve(p, (size_t)BATCH * XDIM * 2);
  _Float16* h1h  = (_Float16*)carve(p, (size_t)BATCH * H1DIM * 2);
  _Float16* h2h  = (_Float16*)carve(p, (size_t)BATCH * H1DIM * 2);
  _Float16* h3h  = (_Float16*)carve(p, (size_t)BATCH * H1DIM * 2);
  _Float16* h4h  = (_Float16*)carve(p, (size_t)BATCH * H2DIM * 2);
  _Float16* W1h  = (_Float16*)carve(p, (size_t)H1DIM * XDIM * 2);
  _Float16* W2h  = (_Float16*)carve(p, (size_t)H1DIM * H1DIM * 2);
  _Float16* W3h  = (_Float16*)carve(p, (size_t)H1DIM * H1DIM * 2);
  _Float16* W4h  = (_Float16*)carve(p, (size_t)H2DIM * H1DIM * 2);

  // Stage 1: bandwidth-bound similarity stream.
  sims_kernel<<<BATCH * 16, 256, 0, stream>>>(stance, body, sims);

  // Weight conversion (independent of stage 1; 20MB read, trivial).
  {
    int n1 = H1DIM * XDIM, n2 = H1DIM * H1DIM, n4 = H2DIM * H1DIM;
    f32_to_f16_kernel<<<(n1 + 255) / 256, 256, 0, stream>>>(W1, W1h, n1);
    f32_to_f16_kernel<<<(n2 + 255) / 256, 256, 0, stream>>>(W2, W2h, n2);
    f32_to_f16_kernel<<<(n2 + 255) / 256, 256, 0, stream>>>(W3, W3h, n2);
    f32_to_f16_kernel<<<(n4 + 255) / 256, 256, 0, stream>>>(W4, W4h, n4);
  }

  // Stage 2: top-5 + concat into f16 activations.
  topk_concat_kernel<<<BATCH, 256, 0, stream>>>(stance, body, sims, xh);

  // Stage 3: WMMA MLP. grid = (Nout/16) blocks of 8 waves (8 M-tiles each).
  wmma_linear_kernel<<<H1DIM / 16, 256, 0, stream>>>(xh,  W1h, b1, h1h, BATCH, XDIM,  H1DIM, 1);
  wmma_linear_kernel<<<H1DIM / 16, 256, 0, stream>>>(h1h, W2h, b2, h2h, BATCH, H1DIM, H1DIM, 1);
  wmma_linear_kernel<<<H1DIM / 16, 256, 0, stream>>>(h2h, W3h, b3, h3h, BATCH, H1DIM, H1DIM, 1);
  wmma_linear_kernel<<<H2DIM / 16, 256, 0, stream>>>(h3h, W4h, b4, h4h, BATCH, H1DIM, H2DIM, 1);

  // Final 512 -> 2 logits in fp32.
  final_layer_kernel<<<(BATCH * CDIM * 32) / 256, 256, 0, stream>>>(h4h, W5, b5, out);
}
